// STIXBertEncoder_70068096467707
// MI455X (gfx1250) — compile-verified, matrix operand order checked by reading, not run on hardware
//
#include <hip/hip_runtime.h>
#include <cmath>

// ---------------------------------------------------------------------------
// Types for gfx1250 WMMA (wave32): v16h = 8 VGPRs of packed f16, v8f = 8 VGPRs
// ---------------------------------------------------------------------------
typedef _Float16 v16h __attribute__((ext_vector_type(16)));
typedef _Float16 v8h  __attribute__((ext_vector_type(8)));
typedef float    v8f  __attribute__((ext_vector_type(8)));

__device__ inline v8f wmma16(v16h a, v16h b, v8f c) {
  // D = A(16x32 f16) * B(32x16 f16) + C(16x16 f32)
  return __builtin_amdgcn_wmma_f32_16x16x32_f16(false, a, false, b, (short)0, c,
                                                false, false);
}

// CDNA5 async memory->LDS copy (no VGPR data path, tracked by ASYNCcnt).
// GV addressing: global_load_async_to_lds_b128 vdst(lds addr), vaddr64, off
__device__ inline void async_load_lds_b128(unsigned lds_byte_addr,
                                           const void* gaddr) {
  asm volatile("global_load_async_to_lds_b128 %0, %1, off"
               :
               : "v"(lds_byte_addr), "v"(gaddr)
               : "memory");
}
__device__ inline void wait_asynccnt0() {
  asm volatile("s_wait_asynccnt 0x0" ::: "memory");
}

__device__ inline void atomic_max_f32(float* addr, float val) {
  // sign-magnitude monotonic trick; buffer initialized to -inf
  if (val >= 0.0f) atomicMax((int*)addr, __float_as_int(val));
  else             atomicMin((unsigned int*)addr, __float_as_uint(val));
}

// ---------------------------------------------------------------------------
// Weight pre-pack: row-major f32 W[K][N] -> f16 WMMA B-fragment layout.
// frag f = nt*(K/32)+kt, within frag: lane*16 + e,
// element (lane,e) <- W[k][n], n = nt*16 + (lane&15),
//                    k = kt*32 + (lane>>4)*8 + (e<8 ? e : e+8)
// ---------------------------------------------------------------------------
__global__ void k_pack_b(const float* __restrict__ src, _Float16* __restrict__ dst,
                         int K, int N, long total) {
  long idx = (long)blockIdx.x * 256 + threadIdx.x;
  if (idx >= total) return;
  int  mk = K * N;
  long m  = idx / mk;
  int  r  = (int)(idx - m * mk);
  int  e    = r & 15;
  int  lane = (r >> 4) & 31;
  int  rest = r >> 9;
  int  ktiles = K >> 5;
  int  kt = rest % ktiles;
  int  nt = rest / ktiles;
  int  n  = (nt << 4) + (lane & 15);
  int  kk = ((lane >> 4) << 3) + (e < 8 ? e : e + 8);
  int  k  = (kt << 5) + kk;
  dst[idx] = (_Float16)src[m * (long)mk + (long)k * N + n];
}

// ---------------------------------------------------------------------------
// Elementwise helpers
// ---------------------------------------------------------------------------
__global__ void k_f32_to_f16(const float* __restrict__ s, _Float16* __restrict__ d, long n) {
  long i = (long)blockIdx.x * 256 + threadIdx.x;
  if (i < n) d[i] = (_Float16)s[i];
}

__global__ void k_fill_f32(float* __restrict__ p, float v, long n) {
  long i = (long)blockIdx.x * 256 + threadIdx.x;
  if (i < n) p[i] = v;
}

__global__ void k_gelu_to_f16(const float* __restrict__ s, _Float16* __restrict__ d, long n) {
  long i = (long)blockIdx.x * 256 + threadIdx.x;
  if (i < n) {
    float x = s[i];
    float t = tanhf(0.7978845608028654f * (x + 0.044715f * x * x * x));
    d[i] = (_Float16)(0.5f * x * (1.0f + t));
  }
}

// ---------------------------------------------------------------------------
// Main GEMM: C[M,256] = A_f16[M,256] @ W(packed f16)[256,256] + bias
// Block = 256 thr = 8 waves (4 in M x 2 in N); block tile 128x64;
// wave tile 32x32 (2x2 WMMA tiles).
// A tiles staged with CDNA5 async memory->LDS copies (ASYNCcnt), double
// buffered; B fragments read from L2-resident packed global memory with
// prefetch hints.
// ---------------------------------------------------------------------------
__global__ __launch_bounds__(256) void k_gemm256(
    const _Float16* __restrict__ A,    // [M,256] row-major f16
    const _Float16* __restrict__ Bp,   // packed [16 ntile][8 ktile][512]
    const float*    __restrict__ bias, // [256]
    float*          __restrict__ Cf,   // optional f32 out [M,256]
    _Float16*       __restrict__ Ch,   // optional f16 out [M,256]
    int M) {
  __shared__ _Float16 As[2][128 * 32];

  const int tid  = threadIdx.x;
  const int lane = tid & 31;
  const int wave = tid >> 5;
  const int wm   = wave >> 1;          // 0..3
  const int wn   = wave & 1;           // 0..1
  const int r0   = blockIdx.x * 128;
  const int n0   = blockIdx.y * 64;

  // LDS byte address of As[0][0] (flat LDS address low 32 bits == LDS offset)
  const unsigned ldsBase = (unsigned)(size_t)(&As[0][0]);

  // this thread's two 16B staging chunks: chunk ch covers row=ch>>2, 8 halves
  const int chRow0 = (tid << 1) >> 2;         // row of chunk 0
  const int chKc0  = ((tid << 1) & 3) << 3;   // k-col of chunk 0
  const int chRow1 = ((tid << 1) + 1) >> 2;
  const int chKc1  = (((tid << 1) + 1) & 3) << 3;

  v8f acc[2][2] = {};

  // prologue: async-stage A tile kt=0 into buffer 0
  async_load_lds_b128(ldsBase + (unsigned)(chRow0 * 32 + chKc0) * 2,
                      A + (size_t)(r0 + chRow0) * 256 + chKc0);
  async_load_lds_b128(ldsBase + (unsigned)(chRow1 * 32 + chKc1) * 2,
                      A + (size_t)(r0 + chRow1) * 256 + chKc1);

  for (int kt = 0; kt < 8; ++kt) {
    wait_asynccnt0();     // my async LDS writes for buffer kt&1 are complete
    __syncthreads();      // everyone's are
    if (kt < 7) {
      const int k0n = (kt + 1) << 5;
      const unsigned bufOfs = (unsigned)(((kt + 1) & 1) * (128 * 32) * 2);
      async_load_lds_b128(ldsBase + bufOfs + (unsigned)(chRow0 * 32 + chKc0) * 2,
                          A + (size_t)(r0 + chRow0) * 256 + k0n + chKc0);
      async_load_lds_b128(ldsBase + bufOfs + (unsigned)(chRow1 * 32 + chKc1) * 2,
                          A + (size_t)(r0 + chRow1) * 256 + k0n + chKc1);
      // prefetch next K-step's B fragments (L2-resident packed weights)
      int ntp = (blockIdx.y << 2) + (wn << 1);
      __builtin_prefetch(Bp + (((size_t)ntp * 8 + kt + 1) << 9) + (lane << 4), 0, 1);
    }
    const int buf   = kt & 1;
    const int kbase = (lane >> 4) << 3;

    v16h a[2];
#pragma unroll
    for (int ms = 0; ms < 2; ++ms) {
      int row = (wm << 5) + (ms << 4) + (lane & 15);
      v8h lo = *(const v8h*)&As[buf][row * 32 + kbase];
      v8h hi = *(const v8h*)&As[buf][row * 32 + kbase + 16];
      a[ms] = __builtin_shufflevector(lo, hi, 0, 1, 2, 3, 4, 5, 6, 7, 8, 9, 10,
                                      11, 12, 13, 14, 15);
    }
#pragma unroll
    for (int ns = 0; ns < 2; ++ns) {
      int ntg = (blockIdx.y << 2) + (wn << 1) + ns;   // global n-tile 0..15
      v16h b = *(const v16h*)(Bp + (((size_t)ntg * 8 + kt) << 9) + (lane << 4));
      acc[0][ns] = wmma16(a[0], b, acc[0][ns]);
      acc[1][ns] = wmma16(a[1], b, acc[1][ns]);
    }
  }

  // epilogue: C layout per ISA: col = lane&15, row = (lane>>4)*8 + r
#pragma unroll
  for (int ms = 0; ms < 2; ++ms) {
#pragma unroll
    for (int ns = 0; ns < 2; ++ns) {
      int col   = n0 + (wn << 5) + (ns << 4) + (lane & 15);
      int rbase = r0 + (wm << 5) + (ms << 4) + ((lane >> 4) << 3);
      float bv  = bias ? bias[col] : 0.0f;
#pragma unroll
      for (int rr = 0; rr < 8; ++rr) {
        float v = acc[ms][ns][rr] + bv;
        size_t o = (size_t)(rbase + rr) * 256 + col;
        if (Cf) Cf[o] = v;
        if (Ch) Ch[o] = (_Float16)v;
      }
    }
  }
}

// ---------------------------------------------------------------------------
// Per-head relation transform: Out[:,h*32:h*32+32] = A[:,h*32:+32] @ Rel_h(32x32)
// K = 32 -> exactly one f16 WMMA per 16x16 tile. grid = (M/128, H)
// ---------------------------------------------------------------------------
__global__ __launch_bounds__(256) void k_rel32(
    const _Float16* __restrict__ A,    // [M,256] f16
    const _Float16* __restrict__ Bp,   // packed, + h*1024 per head
    _Float16*       __restrict__ Out,  // [M,256] f16
    int M) {
  const int tid  = threadIdx.x;
  const int lane = tid & 31;
  const int wave = tid >> 5;
  const int h    = blockIdx.y;
  const int row0 = blockIdx.x * 128 + wave * 16;

  const int laneRow = row0 + (lane & 15);
  const int kbase   = (lane >> 4) << 3;
  const _Float16* ar = A + (size_t)laneRow * 256 + h * 32 + kbase;
  v8h lo = *(const v8h*)(ar);
  v8h hi = *(const v8h*)(ar + 16);
  v16h a = __builtin_shufflevector(lo, hi, 0, 1, 2, 3, 4, 5, 6, 7, 8, 9, 10, 11,
                                   12, 13, 14, 15);
  const _Float16* bb = Bp + (size_t)h * 1024;
#pragma unroll
  for (int ns = 0; ns < 2; ++ns) {
    v16h b = *(const v16h*)(bb + ns * 512 + (lane << 4));
    v8f acc = {};
    acc = wmma16(a, b, acc);
    int col   = h * 32 + (ns << 4) + (lane & 15);
    int rbase = row0 + ((lane >> 4) << 3);
#pragma unroll
    for (int rr = 0; rr < 8; ++rr)
      Out[(size_t)(rbase + rr) * 256 + col] = (_Float16)acc[rr];
  }
}

// ---------------------------------------------------------------------------
// Edge phase: score -> segment max (atomic) -> exp/sum (atomic) -> aggregate
// One thread per (edge, head). E*H = 524288 threads.
// ---------------------------------------------------------------------------
__global__ __launch_bounds__(256) void k_edge_score(
    const int* __restrict__ ei,            // [2,E]
    const _Float16* __restrict__ q16,      // dst-type q, [N,256]
    const _Float16* __restrict__ kr16,     // relation key, [N,256]
    const float* __restrict__ prel,        // [H]
    float* __restrict__ sc,                // [E,H]
    float* __restrict__ mx,                // [N,H]
    int E) {
  int idx = blockIdx.x * 256 + threadIdx.x;
  if (idx >= E * 8) return;
  int e = idx >> 3, h = idx & 7;
  int src = ei[e], dst = ei[E + e];
  const v8h* qv = (const v8h*)(q16 + (size_t)dst * 256 + h * 32);
  const v8h* kv = (const v8h*)(kr16 + (size_t)src * 256 + h * 32);
  float s = 0.0f;
#pragma unroll
  for (int c = 0; c < 4; ++c) {
    v8h qa = qv[c], ka = kv[c];
#pragma unroll
    for (int j = 0; j < 8; ++j) s += (float)qa[j] * (float)ka[j];
  }
  s *= prel[h] * 0.17677669529663687f;  // 1/sqrt(32)
  sc[idx] = s;
  atomic_max_f32(mx + (size_t)dst * 8 + h, s);
}

__global__ __launch_bounds__(256) void k_edge_exp(
    const int* __restrict__ ei, float* __restrict__ sc,
    const float* __restrict__ mx, float* __restrict__ den, int E) {
  int idx = blockIdx.x * 256 + threadIdx.x;
  if (idx >= E * 8) return;
  int e = idx >> 3, h = idx & 7;
  int dst = ei[E + e];
  float ex = __expf(sc[idx] - mx[(size_t)dst * 8 + h]);
  sc[idx] = ex;
  atomicAdd(den + (size_t)dst * 8 + h, ex);
}

__global__ __launch_bounds__(256) void k_edge_agg(
    const int* __restrict__ ei, const float* __restrict__ sc,
    const float* __restrict__ den, const _Float16* __restrict__ mr16,
    float* __restrict__ agg, int E) {
  int idx = blockIdx.x * 256 + threadIdx.x;
  if (idx >= E * 8) return;
  int e = idx >> 3, h = idx & 7;
  int src = ei[e], dst = ei[E + e];
  float alpha = sc[idx] / fmaxf(den[(size_t)dst * 8 + h], 1e-16f);
  const v8h* mv = (const v8h*)(mr16 + (size_t)src * 256 + h * 32);
  float* ag = agg + (size_t)dst * 256 + h * 32;
#pragma unroll
  for (int c = 0; c < 4; ++c) {
    v8h mm = mv[c];
#pragma unroll
    for (int j = 0; j < 8; ++j) atomicAdd(ag + c * 8 + j, alpha * (float)mm[j]);
  }
}

// ---------------------------------------------------------------------------
// conv = sigmoid(skip)*out + (1-a)*h ; y = conv + h ; LayerNorm(y) -> h, h16
// One 256-thread block per row.
// ---------------------------------------------------------------------------
__global__ __launch_bounds__(256) void k_skip_ln(
    const float* __restrict__ outv, float* __restrict__ h,
    _Float16* __restrict__ h16, const float* __restrict__ skipP,
    const float* __restrict__ g, const float* __restrict__ b) {
  __shared__ float red[256];
  int row = blockIdx.x, i = threadIdx.x;
  size_t o = (size_t)row * 256 + i;
  float ov = outv[o];
  float hv = h[o];
  float a  = 1.0f / (1.0f + __expf(-skipP[0]));
  float y  = a * ov + (1.0f - a) * hv + hv;

  red[i] = y;
  __syncthreads();
  for (int s = 128; s > 0; s >>= 1) {
    if (i < s) red[i] += red[i + s];
    __syncthreads();
  }
  float mu = red[0] * (1.0f / 256.0f);
  __syncthreads();
  float d = y - mu;
  red[i] = d * d;
  __syncthreads();
  for (int s = 128; s > 0; s >>= 1) {
    if (i < s) red[i] += red[i + s];
    __syncthreads();
  }
  float inv = rsqrtf(red[0] * (1.0f / 256.0f) + 1e-5f);
  float res = d * inv * g[i] + b[i];
  h[o]  = res;
  h16[o] = (_Float16)res;
}

// ---------------------------------------------------------------------------
// Host orchestration
// ---------------------------------------------------------------------------
extern "C" void kernel_launch(void* const* d_in, const int* in_sizes, int n_in,
                              void* d_out, int out_size, void* d_ws, size_t ws_size,
                              hipStream_t stream) {
  (void)in_sizes; (void)n_in; (void)out_size; (void)ws_size;
  const int Nn = 16384, E = 65536;
  const size_t NH = (size_t)Nn * 256;
  const int RELsrc[4] = {1, 0, 2, 0};
  const int RELdst[4] = {0, 2, 1, 0};

  const float* x[3]  = {(const float*)d_in[0], (const float*)d_in[1], (const float*)d_in[2]};
  const int*   ei[4] = {(const int*)d_in[3], (const int*)d_in[4], (const int*)d_in[5], (const int*)d_in[6]};
  const float* Win  = (const float*)d_in[7];
  const float* b_in = (const float*)d_in[8];
  const float* Kw   = (const float*)d_in[9];
  const float* Kb   = (const float*)d_in[10];
  const float* Qw   = (const float*)d_in[11];
  const float* Qb   = (const float*)d_in[12];
  const float* Vw   = (const float*)d_in[13];
  const float* Vb   = (const float*)d_in[14];
  const float* Arel = (const float*)d_in[15];
  const float* Mrel = (const float*)d_in[16];
  const float* Prel = (const float*)d_in[17];
  const float* Aw   = (const float*)d_in[18];
  const float* Ab   = (const float*)d_in[19];
  const float* skp  = (const float*)d_in[20];
  const float* ln_g = (const float*)d_in[21];
  const float* ln_b = (const float*)d_in[22];
  const float* Wout = (const float*)d_in[23];
  const float* b_out = (const float*)d_in[24];

  // ---- workspace carve ----
  unsigned char* ws = (unsigned char*)d_ws;
  size_t off = 0;
  auto take = [&](size_t bytes) -> void* {
    void* p = ws + off;
    off += (bytes + 255) & ~(size_t)255;
    return p;
  };
  _Float16* WinP  = (_Float16*)take((size_t)3 * 65536 * 2);
  _Float16* KwP   = (_Float16*)take((size_t)12 * 65536 * 2);
  _Float16* QwP   = (_Float16*)take((size_t)12 * 65536 * 2);
  _Float16* VwP   = (_Float16*)take((size_t)12 * 65536 * 2);
  _Float16* AwP   = (_Float16*)take((size_t)12 * 65536 * 2);
  _Float16* WoutP = (_Float16*)take((size_t)3 * 65536 * 2);
  _Float16* ArelP = (_Float16*)take((size_t)128 * 1024 * 2);
  _Float16* MrelP = (_Float16*)take((size_t)128 * 1024 * 2);
  _Float16* x16   = (_Float16*)take(3 * NH * 2);
  float*    hs    = (float*)take(3 * NH * 4);
  _Float16* h16   = (_Float16*)take(3 * NH * 2);
  _Float16* k16   = (_Float16*)take(3 * NH * 2);
  _Float16* q16   = (_Float16*)take(3 * NH * 2);
  _Float16* v16   = (_Float16*)take(3 * NH * 2);
  _Float16* kr16  = (_Float16*)take(4 * NH * 2);
  _Float16* mr16  = (_Float16*)take(4 * NH * 2);
  float*    sc    = (float*)take((size_t)4 * E * 8 * 4);
  float*    mx    = (float*)take((size_t)3 * Nn * 8 * 4);
  float*    den   = (float*)take((size_t)3 * Nn * 8 * 4);
  float*    agg   = (float*)take(3 * NH * 4);   // also reused as GEMM f32 tmp
  _Float16* g16   = (_Float16*)take(3 * NH * 2);

  const dim3 blk(256);
  const dim3 gGemm(Nn / 128, 4);
  const dim3 gRel(Nn / 128, 8);
  const int  gEdge = (E * 8) / 256;

  // ---- pack all weights to WMMA B-fragment layout (f16) ----
  k_pack_b<<<dim3((3 * 65536 + 255) / 256), blk, 0, stream>>>(Win, WinP, 256, 256, (long)3 * 65536);
  k_pack_b<<<dim3((12 * 65536 + 255) / 256), blk, 0, stream>>>(Kw, KwP, 256, 256, (long)12 * 65536);
  k_pack_b<<<dim3((12 * 65536 + 255) / 256), blk, 0, stream>>>(Qw, QwP, 256, 256, (long)12 * 65536);
  k_pack_b<<<dim3((12 * 65536 + 255) / 256), blk, 0, stream>>>(Vw, VwP, 256, 256, (long)12 * 65536);
  k_pack_b<<<dim3((12 * 65536 + 255) / 256), blk, 0, stream>>>(Aw, AwP, 256, 256, (long)12 * 65536);
  k_pack_b<<<dim3((3 * 65536 + 255) / 256), blk, 0, stream>>>(Wout, WoutP, 256, 256, (long)3 * 65536);
  k_pack_b<<<dim3((128 * 1024 + 255) / 256), blk, 0, stream>>>(Arel, ArelP, 32, 32, (long)128 * 1024);
  k_pack_b<<<dim3((128 * 1024 + 255) / 256), blk, 0, stream>>>(Mrel, MrelP, 32, 32, (long)128 * 1024);

  // ---- input projections: h[t] = x[t] @ Win[t] + b_in[t] ----
  for (int t = 0; t < 3; ++t)
    k_f32_to_f16<<<dim3(NH / 256), blk, 0, stream>>>(x[t], x16 + t * NH, (long)NH);
  for (int t = 0; t < 3; ++t)
    k_gemm256<<<gGemm, blk, 0, stream>>>(x16 + t * NH, WinP + (size_t)t * 65536,
                                         b_in + t * 256, hs + t * NH, h16 + t * NH, Nn);

  // ---- layers ----
  for (int l = 0; l < 4; ++l) {
    for (int t = 0; t < 3; ++t) {
      size_t wofs = (size_t)(l * 3 + t) * 65536;
      k_gemm256<<<gGemm, blk, 0, stream>>>(h16 + t * NH, KwP + wofs,
                                           Kb + (l * 3 + t) * 256, nullptr, k16 + t * NH, Nn);
      k_gemm256<<<gGemm, blk, 0, stream>>>(h16 + t * NH, QwP + wofs,
                                           Qb + (l * 3 + t) * 256, nullptr, q16 + t * NH, Nn);
      k_gemm256<<<gGemm, blk, 0, stream>>>(h16 + t * NH, VwP + wofs,
                                           Vb + (l * 3 + t) * 256, nullptr, v16 + t * NH, Nn);
    }
    for (int r = 0; r < 4; ++r) {
      int s = RELsrc[r];
      size_t rofs = (size_t)(l * 4 + r) * 8 * 1024;
      k_rel32<<<gRel, blk, 0, stream>>>(k16 + s * NH, ArelP + rofs, kr16 + r * NH, Nn);
      k_rel32<<<gRel, blk, 0, stream>>>(v16 + s * NH, MrelP + rofs, mr16 + r * NH, Nn);
    }
    k_fill_f32<<<dim3((3 * Nn * 8) / 256), blk, 0, stream>>>(mx, -INFINITY, (long)3 * Nn * 8);
    k_fill_f32<<<dim3((3 * Nn * 8) / 256), blk, 0, stream>>>(den, 0.0f, (long)3 * Nn * 8);
    k_fill_f32<<<dim3((3 * NH) / 256), blk, 0, stream>>>(agg, 0.0f, (long)(3 * NH));

    for (int r = 0; r < 4; ++r) {
      int dt = RELdst[r];
      k_edge_score<<<dim3(gEdge), blk, 0, stream>>>(
          ei[r], q16 + dt * NH, kr16 + r * NH, Prel + (l * 4 + r) * 8,
          sc + (size_t)r * E * 8, mx + (size_t)dt * Nn * 8, E);
    }
    for (int r = 0; r < 4; ++r) {
      int dt = RELdst[r];
      k_edge_exp<<<dim3(gEdge), blk, 0, stream>>>(
          ei[r], sc + (size_t)r * E * 8, mx + (size_t)dt * Nn * 8,
          den + (size_t)dt * Nn * 8, E);
    }
    for (int r = 0; r < 4; ++r) {
      int dt = RELdst[r];
      k_edge_agg<<<dim3(gEdge), blk, 0, stream>>>(
          ei[r], sc + (size_t)r * E * 8, den + (size_t)dt * Nn * 8,
          mr16 + r * NH, agg + (size_t)dt * NH, E);
    }

    k_gelu_to_f16<<<dim3((3 * NH) / 256), blk, 0, stream>>>(agg, g16, (long)(3 * NH));
    for (int t = 0; t < 3; ++t) {
      size_t wofs = (size_t)(l * 3 + t) * 65536;
      // out-linear: f32 result overwrites agg (safe: agg no longer read)
      k_gemm256<<<gGemm, blk, 0, stream>>>(g16 + t * NH, AwP + wofs,
                                           Ab + (l * 3 + t) * 256, agg + t * NH,
                                           nullptr, Nn);
    }
    for (int t = 0; t < 3; ++t) {
      k_skip_ln<<<dim3(Nn), blk, 0, stream>>>(agg + t * NH, hs + t * NH,
                                              h16 + t * NH, skp + l * 3 + t,
                                              ln_g + (size_t)(l * 3 + t) * 256,
                                              ln_b + (size_t)(l * 3 + t) * 256);
    }
  }

  // ---- output projections into d_out (3 x [N,256] f32, concatenated) ----
  for (int t = 0; t < 3; ++t)
    k_gemm256<<<gGemm, blk, 0, stream>>>(h16 + t * NH, WoutP + (size_t)t * 65536,
                                         b_out + t * 256,
                                         (float*)d_out + t * NH, nullptr, Nn);
}
